// My_deconv2d_609885356905
// MI455X (gfx1250) — matrix operand order repaired and checked by reading, not compiled, not run on hardware
//
#include <hip/hip_runtime.h>

// ---------------- problem constants (from reference) ----------------
#define N_N     16384
#define L_N     39
#define N_SUB   8192
#define G_N     2048            // N_SUB / 4
#define TD_FULL (N_N + N_SUB)   // 24576
#define R_N     (G_N + N_SUB)   // 10240 (multiple of 256)
#define KEEPC   22              // cols [0..4] + [22..38]
#define P_N     2048
#define OUT_CH  64
#define OUTW    65              // 1 id col + 64 channels

typedef __attribute__((ext_vector_type(16))) _Float16     v16h;
typedef __attribute__((ext_vector_type(8)))  float        v8f;
typedef __attribute__((ext_vector_type(2)))  float        v2f;
typedef __attribute__((ext_vector_type(4)))  unsigned int u32x4;
typedef __attribute__((ext_vector_type(8)))  int          i32x8;
typedef __attribute__((ext_vector_type(4)))  int          i32x4;

#if __has_builtin(__builtin_amdgcn_wmma_f32_16x16x4_f32)
#define HAVE_WMMA_F32X4 1
#endif
#if __has_builtin(__builtin_amdgcn_tensor_load_to_lds)
#define HAVE_TDM 1
#endif

// ---------------- workspace layout (floats) ----------------
constexpr size_t OFF_DATA   = 0;                               // sorted nb (16384 x 39)
constexpr size_t OFF_TDFULL = OFF_DATA   + (size_t)N_N * L_N;  // 24576 x 4
constexpr size_t OFF_POOL   = OFF_TDFULL + (size_t)TD_FULL * 4;
constexpr size_t OFF_NEW2   = OFF_POOL   + (size_t)G_N * 4;    // 10240 x 22
constexpr size_t OFF_NEW2S  = OFF_NEW2   + (size_t)R_N * KEEPC;
constexpr size_t OFF_TMP    = OFF_NEW2S  + (size_t)R_N * KEEPC;
constexpr size_t OFF_DATA2  = OFF_TMP    + (size_t)R_N * KEEPC;
constexpr size_t OFF_WANT   = OFF_DATA2  + (size_t)R_N * KEEPC;
constexpr size_t OFF_SEL    = OFF_WANT   + (size_t)P_N;
constexpr size_t OFF_END_F  = OFF_SEL    + (size_t)P_N * 4;    // then int region

__device__ __forceinline__ int wrap_idx(int i, int n) { int m = i % n; return (m < 0) ? m + n : m; }

// TDM: DMA a strided key column tile (1 x rows elements, element stride = ncols f32)
// from global into LDS at lds_byte. Issued by one wave; caller must barrier after.
__device__ __forceinline__ void tdm_stage_column(const float* gsrc, unsigned lds_byte,
                                                 int ncols, int rows, int tensor_rows) {
#if defined(HAVE_TDM)
  unsigned long long ga = (unsigned long long)(size_t)gsrc;     // tile start (bytes)
  u32x4 g0;
  g0[0] = 1u;                                                   // count=1, user mode
  g0[1] = lds_byte;                                             // lds_addr
  g0[2] = (unsigned)(ga & 0xFFFFFFFFull);                       // global_addr[31:0]
  g0[3] = (unsigned)((ga >> 32) & 0x1FFFFFFull) | (2u << 30);   // addr[56:32] | type=2
  i32x8 g1;
  g1[0] = (int)(2u << 16);                                      // data_size = 4 bytes
  g1[1] = (int)(((unsigned)ncols & 0xFFFFu) << 16);             // tensor_dim0[15:0]
  g1[2] = (int)(((unsigned)tensor_rows & 0xFFFFu) << 16);       // tensor_dim1[15:0]
  g1[3] = (int)(1u << 16);                                      // tile_dim0 = 1
  g1[4] = (int)(unsigned)rows;                                  // tile_dim1 = rows
  g1[5] = ncols;                                                // tensor_dim0_stride[31:0]
  g1[6] = 0;
  g1[7] = 0;
  i32x4 z4 = {0, 0, 0, 0};
#if defined(__clang_major__) && __clang_major__ >= 23
  i32x8 z8 = {0, 0, 0, 0, 0, 0, 0, 0};
  __builtin_amdgcn_tensor_load_to_lds(g0, g1, z4, z4, z8, 0);
#else
  __builtin_amdgcn_tensor_load_to_lds(g0, g1, z4, z4, 0);
#endif
  __builtin_amdgcn_s_wait_tensorcnt(0);
#endif
}

// Stable descending sort by rank counting (reproduces jax.lax.top_k ordering:
// larger key first; ties broken by smaller original index). O(N^2).
// Key tiles staged into LDS by the Tensor Data Mover. nrows % 256 == 0.
__global__ void rank_sort_kernel(const float* __restrict__ src, float* __restrict__ dst,
                                 int nrows, int ncols, int keycol) {
  __shared__ float skey[256];
  int i = blockIdx.x * 256 + threadIdx.x;
  float myk = src[(size_t)i * ncols + keycol];
  int rank = 0;
  int ntiles = nrows / 256;
  for (int t = 0; t < ntiles; ++t) {
#if defined(HAVE_TDM)
    if (threadIdx.x < 32) {           // one wave drives the TDM for the block
      const float* tile = src + (size_t)t * 256 * ncols + keycol;
      tdm_stage_column(tile, (unsigned)(size_t)skey, ncols, 256, nrows - t * 256);
    }
#else
    skey[threadIdx.x] = src[(size_t)(t * 256 + threadIdx.x) * ncols + keycol];
#endif
    if (threadIdx.x == 0 && t + 1 < ntiles)
      __builtin_prefetch(src + (size_t)(t + 1) * 256 * ncols + keycol, 0, 0);
    __syncthreads();
    for (int jj = 0; jj < 256; ++jj) {
      int jg = t * 256 + jj;
      float kj = skey[jj];
      rank += (kj > myk) || ((kj == myk) && (jg < i));
    }
    __syncthreads();
  }
  for (int c = 0; c < ncols; ++c)
    dst[(size_t)rank * ncols + c] = src[(size_t)i * ncols + c];
}

// td_full = concat(traindata, rev) ; rev built from new_nb col4 / original nb col0
__global__ void build_tdfull(const float* __restrict__ td, const float* __restrict__ nb,
                             const float* __restrict__ data, float* __restrict__ tdfull) {
  int i = blockIdx.x * 256 + threadIdx.x;
  if (i >= TD_FULL) return;
  if (i < N_N) {
    for (int c = 0; c < 4; ++c) tdfull[(size_t)i * 4 + c] = td[(size_t)i * 4 + c];
  } else {
    int s = i - N_N;
    const float* newnb = data + (size_t)N_SUB * L_N;          // data[Nn-N_SUB:]
    bool ins = (newnb[(size_t)s * L_N + 4] == 0.f);
    tdfull[(size_t)i * 4 + 0] = ins ? nb[(size_t)s * L_N + 0] : -1e30f;
    tdfull[(size_t)i * 4 + 1] = 0.f;
    tdfull[(size_t)i * 4 + 2] = 0.f;
    tdfull[(size_t)i * 4 + 3] = 0.f;
  }
}

// One wave32 per group g: sequential j=0..3 with wave-parallel first-match scan.
__global__ void group_select(const float* __restrict__ data, const float* __restrict__ tdfull,
                             float* __restrict__ pool, int* __restrict__ nbmax_out) {
  int lane = threadIdx.x & 31;
  int g = (blockIdx.x * blockDim.x + threadIdx.x) >> 5;
  if (g >= G_N) return;
  const float* newnb = data + (size_t)N_SUB * L_N;
  float maxmin = -100000.f;
  int maxidx = -100, nbmax = -100;
  for (int j = 0; j < 4; ++j) {
    float target = newnb[(size_t)(4 * g + j) * L_N + 0];
    int mn = 0x7fffffff;
    for (int t = lane; t < TD_FULL; t += 32)
      if (tdfull[(size_t)t * 4] == target) { mn = t; break; }   // ascending per lane
    for (int off = 16; off; off >>= 1) {
      int o = __shfl_xor(mn, off, 32);
      mn = (o < mn) ? o : mn;
    }
    int tp = (mn == 0x7fffffff) ? 0 : mn;                        // argmax-no-match -> 0
    float a = tdfull[(size_t)tp * 4 + 1];
    float b = tdfull[(size_t)tp * 4 + 2];
    float c = tdfull[(size_t)tp * 4 + 3];
    float mx = fmaxf(a, fmaxf(b, c));
    bool flag = (newnb[(size_t)(4 * g + j) * L_N + 4] != 0.f);
    bool upd = (flag == (mx > maxmin));
    if (upd) { maxmin = mx; maxidx = tp; nbmax = 4 * g + j; }
  }
  if (lane == 0) {
    int mi = wrap_idx(maxidx, TD_FULL);                          // JAX negative-index wrap
    for (int c = 0; c < 4; ++c) pool[(size_t)g * 4 + c] = tdfull[(size_t)mi * 4 + c];
    nbmax_out[g] = nbmax;
  }
}

// new2 = concat(new_nb[nbmax], shengyu)[:, KEEP]
__global__ void build_new2(const float* __restrict__ data, const int* __restrict__ nbmax,
                           float* __restrict__ new2) {
  int r = blockIdx.x * 256 + threadIdx.x;
  if (r >= R_N) return;
  const float* src;
  if (r < G_N) {
    int idx = wrap_idx(nbmax[r], N_SUB);
    src = data + (size_t)(N_SUB + idx) * L_N;
  } else {
    src = data + (size_t)(r - G_N) * L_N;                        // shengyu = data[:8192]
  }
  for (int c = 0; c < 5; ++c)  new2[(size_t)r * KEEPC + c]     = src[c];
  for (int v = 0; v < 17; ++v) new2[(size_t)r * KEEPC + 5 + v] = src[22 + v];
}

// len_s = count(new2 col3 == -1); dl = count(col2 == -1) (sort-invariant multisets)
__global__ void count_flags(const float* __restrict__ new2, int* __restrict__ ctrs) {
  int i = blockIdx.x * 256 + threadIdx.x;
  if (i >= R_N) return;
  if (new2[(size_t)i * KEEPC + 3] == -1.f) atomicAdd(&ctrs[0], 1);
  if (new2[(size_t)i * KEEPC + 2] == -1.f) atomicAdd(&ctrs[1], 1);
}

// tmp = where(row < len_s, new2s, td_rows); one wave per row, 17 first-match scans.
// The 10240-entry match column (col 3) is staged once per block into LDS with
// async global->LDS loads (ASYNCcnt), then all scans run out of LDS.
__global__ void build_tmp(const float* __restrict__ n2s, float* __restrict__ tmp,
                          const int* __restrict__ ctrs) {
  __shared__ float scol3[R_N];                                   // 40 KB (WGP has 320 KB)
  for (int t = threadIdx.x; t < R_N; t += 256) {
    unsigned lds = (unsigned)(size_t)&scol3[t];
    unsigned long long ga = (unsigned long long)(size_t)(n2s + (size_t)t * KEEPC + 3);
    asm volatile("global_load_async_to_lds_b32 %0, %1, off"
                 :: "v"(lds), "v"(ga) : "memory");
  }
  asm volatile("s_wait_asynccnt 0x0" ::: "memory");
  __syncthreads();

  int lane = threadIdx.x & 31;
  int r = (blockIdx.x * blockDim.x + threadIdx.x) >> 5;
  if (r >= R_N) return;
  int len_s = ctrs[0];
  if (r < len_s) {
    for (int c = lane; c < KEEPC; c += 32)
      tmp[(size_t)r * KEEPC + c] = n2s[(size_t)r * KEEPC + c];
    return;
  }
  float vals[17];
  for (int v = 0; v < 17; ++v) {
    float ref = n2s[(size_t)r * KEEPC + 5 + v];
    float val = ref;
    if (ref > 0.f) {
      int mn = 0x7fffffff;
      for (int t = lane; t < R_N; t += 32)
        if (scol3[t] == ref) { mn = t; break; }
      for (int off = 16; off; off >>= 1) {
        int o = __shfl_xor(mn, off, 32);
        mn = (o < mn) ? o : mn;
      }
      int p = (mn == 0x7fffffff) ? 0 : mn;
      val = n2s[(size_t)p * KEEPC + 0];
    }
    vals[v] = val;
  }
  if (lane == 0) {
    for (int c = 0; c < 5; ++c)  tmp[(size_t)r * KEEPC + c]     = n2s[(size_t)r * KEEPC + c];
    for (int v = 0; v < 17; ++v) tmp[(size_t)r * KEEPC + 5 + v] = vals[v];
  }
}

__global__ void build_want(const float* __restrict__ data2, const int* __restrict__ ctrs,
                           float* __restrict__ want) {
  int k = blockIdx.x * 256 + threadIdx.x;
  if (k >= P_N) return;
  int dl = ctrs[1];
  int idx = dl + k; if (idx > R_N - 1) idx = R_N - 1;
  want[k] = data2[(size_t)idx * KEEPC + 0];
}

__global__ void build_sel(const float* __restrict__ want, const float* __restrict__ pool,
                          float* __restrict__ sel) {
  int k = blockIdx.x * 256 + threadIdx.x;
  if (k >= P_N) return;
  float w = want[k];
  int p = 0;
  for (int j = 0; j < P_N; ++j)
    if (pool[(size_t)j * 4] == w) { p = j; break; }               // first match, else 0
  for (int c = 0; c < 4; ++c) sel[(size_t)k * 4 + c] = pool[(size_t)p * 4 + c];
}

// Final: tl = win(8192x12) * kernel^T(12x64) via WMMA; out = [id | tl] (8192x65).
// win row i has a single nonzero 4-row window slot f_i = (3 - i%4)&3, data = sel[k_i][1:4].
__global__ void wmma_out_kernel(const float* __restrict__ sel, const float* __restrict__ ker,
                                const float* __restrict__ data, float* __restrict__ out) {
  int lane = threadIdx.x & 31;
  int half = lane >> 4;
  int m = lane & 15;
  int m0 = blockIdx.x * 16;
  int i = m0 + m;
  int fi = (3 - (i & 3)) & 3;
  int ki = (i + fi - 3) >> 2;
  float s0 = sel[(size_t)ki * 4 + 1];
  float s1 = sel[(size_t)ki * 4 + 2];
  float s2 = sel[(size_t)ki * 4 + 3];
  if (half == 0) out[(size_t)i * OUTW] = data[(size_t)(N_SUB + i) * L_N]; // lastnb col0

  for (int nt = 0; nt < 4; ++nt) {
    v8f c = {};
#if defined(HAVE_WMMA_F32X4)
    // exact f32: three chained 16x16x4 f32 WMMAs cover K=0..11
    for (int ks = 0; ks < 3; ++ks) {
      v2f A, B;
      for (int e = 0; e < 2; ++e) {
        int K = ks * 4 + half * 2 + e;                 // ISA 16x4 f32 A layout
        float av = 0.f;
        if (K < 12 && (K / 3) == fi) av = (K % 3 == 0) ? s0 : ((K % 3 == 1) ? s1 : s2);
        A[e] = av;
        int o = nt * 16 + m;                           // B column = lane%16
        B[e] = (K < 12) ? ker[(size_t)o * 12 + K] : 0.f;
      }
      c = __builtin_amdgcn_wmma_f32_16x16x4_f32(false, A, false, B, (short)0, c, false, false);
    }
#else
    v16h A, B;
    for (int e = 0; e < 16; ++e) {
      int Ka = ((e < 8) ? e : e + 8) + (half ? 8 : 0); // ISA 16-bit 16x32 A layout
      float av = 0.f;
      if (Ka < 12 && (Ka / 3) == fi) av = (Ka % 3 == 0) ? s0 : ((Ka % 3 == 1) ? s1 : s2);
      A[e] = (_Float16)av;
      int Kb = ((e < 8) ? e : e + 8) + (half ? 8 : 0);
      int o = nt * 16 + m;
      B[e] = (_Float16)((Kb < 12) ? ker[(size_t)o * 12 + Kb] : 0.f);
    }
    c = __builtin_amdgcn_wmma_f32_16x16x32_f16(false, A, false, B, (short)0, c, false, false);
#endif
    for (int v = 0; v < 8; ++v) {                      // ISA 16x16 f32 C/D layout
      int row = m0 + v + half * 8;
      int col = nt * 16 + m;
      out[(size_t)row * OUTW + 1 + col] = c[v];
    }
  }
}

extern "C" void kernel_launch(void* const* d_in, const int* in_sizes, int n_in,
                              void* d_out, int out_size, void* d_ws, size_t ws_size,
                              hipStream_t stream) {
  const float* td  = (const float*)d_in[0];   // traindata (16384,4)
  const float* nb  = (const float*)d_in[1];   // neighbor  (1,16384,39)
  const float* ker = (const float*)d_in[2];   // kernel    (64,4,3)
  float* ws     = (float*)d_ws;
  float* data   = ws + OFF_DATA;
  float* tdfull = ws + OFF_TDFULL;
  float* pool   = ws + OFF_POOL;
  float* new2   = ws + OFF_NEW2;
  float* new2s  = ws + OFF_NEW2S;
  float* tmp    = ws + OFF_TMP;
  float* data2  = ws + OFF_DATA2;
  float* want   = ws + OFF_WANT;
  float* sel    = ws + OFF_SEL;
  int*   iws    = (int*)(ws + OFF_END_F);
  int*   nbmax  = iws;                        // [G_N]
  int*   ctrs   = iws + G_N;                  // [len_s, dl]
  float* out    = (float*)d_out;

  hipMemsetAsync(ctrs, 0, 2 * sizeof(int), stream);

  rank_sort_kernel<<<N_N / 256, 256, 0, stream>>>(nb, data, N_N, L_N, 1);        // sort by M_
  build_tdfull    <<<(TD_FULL + 255) / 256, 256, 0, stream>>>(td, nb, data, tdfull);
  group_select    <<<(G_N * 32) / 256, 256, 0, stream>>>(data, tdfull, pool, nbmax);
  build_new2      <<<(R_N + 255) / 256, 256, 0, stream>>>(data, nbmax, new2);
  count_flags     <<<(R_N + 255) / 256, 256, 0, stream>>>(new2, ctrs);
  rank_sort_kernel<<<R_N / 256, 256, 0, stream>>>(new2, new2s, R_N, KEEPC, 2);   // sort by M
  build_tmp       <<<(R_N * 32) / 256, 256, 0, stream>>>(new2s, tmp, ctrs);
  rank_sort_kernel<<<R_N / 256, 256, 0, stream>>>(tmp, data2, R_N, KEEPC, 1);    // sort by M_
  build_want      <<<(P_N + 255) / 256, 256, 0, stream>>>(data2, ctrs, want);
  build_sel       <<<(P_N + 255) / 256, 256, 0, stream>>>(want, pool, sel);
  wmma_out_kernel <<<N_SUB / 16, 32, 0, stream>>>(sel, ker, data, out);
}
